// BayerFeatureExtractor_73924977098936
// MI455X (gfx1250) — matrix operand order repaired and checked by reading, not compiled, not run on hardware
//
#include <hip/hip_runtime.h>

#define HW    768
#define NPIX  (HW*HW)
#define NBAT  2
#define EPSF  1e-6f

// workspace layout (in floats)
#define WB0_OFF 0        // [16][32]  bank0 weights (phase-independent)
#define WB1_OFF 512      // [4][16][32] bank1 weights per Bayer phase
#define DEN_OFF 2560     // [4][5]  smooth5*mask normalizers per phase
#define FLD_OFF 4096     // [6][NBAT][HW][HW]: mhcf, numR,numG,numB,numGR,numGB

typedef __attribute__((ext_vector_type(16))) __bf16 v16bf;
typedef __attribute__((ext_vector_type(8)))  float  v8f;

__device__ __forceinline__ int refl(int i, int n) {
  if (i < 0) i = -i;
  if (i >= n) i = 2*n - 2 - i;
  return i;
}

__device__ __forceinline__ float maskv(int c, int ry, int rx) {
  // gbrg: gb(0,0) b(0,1) r(1,0) gr(1,1)
  switch (c) {
    case 0:  return (ry==1 && rx==0) ? 1.f : 0.f;  // r
    case 1:  return (ry==rx)         ? 1.f : 0.f;  // g
    case 2:  return (ry==0 && rx==1) ? 1.f : 0.f;  // b
    case 3:  return (ry==1 && rx==1) ? 1.f : 0.f;  // gr
    default: return (ry==0 && rx==0) ? 1.f : 0.f;  // gb
  }
}

// ---------------------------------------------------------------------------
// K0: build bf16-WMMA weight tables (needs expf/cosf -> runtime, 1 thread)
// ---------------------------------------------------------------------------
__global__ void bf_init(float* ws) {
  if (threadIdx.x != 0 || blockIdx.x != 0) return;
  float k5[19][25];
  for (int i = 0; i < 19; ++i)
    for (int t = 0; t < 25; ++t) k5[i][t] = 0.f;

  const float K3D[9][9] = {
    {0,1,0, 1,-4,1, 0,1,0},      // lap
    {0,0,0, 1,-2,1, 0,0,0},      // hxx
    {0,1,0, 0,-2,0, 0,1,0},      // hyy
    {1,0,-1, 0,0,0, -1,0,1},     // hxy
    {-1,0,1, -2,0,2, -1,0,1},    // sobel x
    {-1,-2,-1, 0,0,0, 1,2,1},    // sobel y
    {-2,-1,0, -1,0,1, 0,1,2},    // diag main
    {0,1,2, -1,0,1, -2,-1,0},    // diag anti
    {1,-1,1, -1,1,-1, 1,-1,1},   // checkerboard
  };
  const float K3S[9] = {1.f,1.f,1.f,0.25f,0.125f,0.125f,0.125f,0.125f,1.f/9.f};
  for (int c = 0; c < 9; ++c)
    for (int dy = 0; dy < 3; ++dy)
      for (int dx = 0; dx < 3; ++dx)
        k5[c][(dy+1)*5 + (dx+1)] = K3D[c][dy*3+dx] * K3S[c];

  const float hs[5] = {-0.5f, 0.f, 1.f, 0.f, -0.5f};
  const float hg[5] = {-0.25f, 0.5f, 0.5f, 0.5f, -0.25f};
  const float st[5] = {0.25f, -1.f, 1.5f, -1.f, 0.25f};
  for (int i = 0; i < 5; ++i) {
    k5[9][2*5+i]  = hs[i];  k5[10][i*5+2] = hs[i];   // ha_second h/v
    k5[11][2*5+i] = hg[i];  k5[12][i*5+2] = hg[i];   // ha_green h/v
    k5[14][2*5+i] = st[i];  k5[15][i*5+2] = st[i];   // stripe x/y
  }
  const float MHC[25] = {0,0,-1,0,0, 0,0,2,0,0, -1,2,4,2,-1, 0,0,2,0,0, 0,0,-1,0,0};
  for (int t = 0; t < 25; ++t) k5[13][t] = MHC[t] * 0.125f;

  const float PI = 3.14159265358979323846f;
  for (int gi = 0; gi < 2; ++gi) {                    // gabor 45 / 135
    float theta = (gi == 0) ? PI*0.25f : 3.f*PI*0.25f;
    float sg = 1.1f, lb = 3.f, gm = 0.65f;
    float ct = cosf(theta), sn = sinf(theta);
    float tmp[25]; float mean = 0.f;
    for (int iy = 0; iy < 5; ++iy)
      for (int ix = 0; ix < 5; ++ix) {
        float yy = (float)(iy-2), xx = (float)(ix-2);
        float xt =  xx*ct + yy*sn;
        float yt = -xx*sn + yy*ct;
        float v = expf(-(xt*xt + gm*gm*yt*yt) / (2.f*sg*sg)) * cosf(2.f*PI*xt/lb);
        tmp[iy*5+ix] = v; mean += v;
      }
    mean *= (1.f/25.f);
    float asum = 0.f;
    for (int t = 0; t < 25; ++t) { tmp[t] -= mean; asum += fabsf(tmp[t]); }
    asum = fmaxf(asum, 1e-6f);
    for (int t = 0; t < 25; ++t) k5[16+gi][t] = tmp[t] / asum;
  }
  {                                                   // dct_like(2,2,5)
    float tmp[25]; float mean = 0.f;
    for (int iy = 0; iy < 5; ++iy)
      for (int ix = 0; ix < 5; ++ix) {
        float v = cosf(PI*(2*ix+1)*2.f/10.f) * cosf(PI*(2*iy+1)*2.f/10.f);
        tmp[iy*5+ix] = v; mean += v;
      }
    mean *= (1.f/25.f);
    float asum = 0.f;
    for (int t = 0; t < 25; ++t) { tmp[t] -= mean; asum += fabsf(tmp[t]); }
    asum = fmaxf(asum, 1e-6f);
    for (int t = 0; t < 25; ++t) k5[18][t] = tmp[t] / asum;
  }

  const float s1[5] = {1,2,3,2,1};
  float sm[25];
  for (int iy = 0; iy < 5; ++iy)
    for (int ix = 0; ix < 5; ++ix) sm[iy*5+ix] = s1[iy]*s1[ix] / 81.f;

  // bank0: conv channels 0..15 (lap..chk, hsh,hsv,hgh,hgv,mhcf,stx,sty)
  for (int m = 0; m < 16; ++m)
    for (int k = 0; k < 32; ++k)
      ws[WB0_OFF + m*32 + k] = (k < 25) ? k5[m][k] : 0.f;

  // bank1 per phase: g45,g135,dct + 5 phase-folded smooth5*mask numerators
  for (int ph = 0; ph < 4; ++ph) {
    int py = ph >> 1, px = ph & 1;
    for (int m = 0; m < 16; ++m)
      for (int k = 0; k < 32; ++k) {
        float v = 0.f;
        if (k < 25) {
          if (m < 3) v = k5[16+m][k];
          else if (m < 8) {
            int c = m - 3;
            int dy = k/5 - 2, dx = k%5 - 2;
            v = sm[k] * maskv(c, (py+dy+4)&1, (px+dx+4)&1);
          }
        }
        ws[WB1_OFF + ph*512 + m*32 + k] = v;
      }
    for (int c = 0; c < 5; ++c) {
      float d = 0.f;
      for (int k = 0; k < 25; ++k) {
        int dy = k/5 - 2, dx = k%5 - 2;
        d += sm[k] * maskv(c, (py+dy+4)&1, (px+dx+4)&1);
      }
      ws[DEN_OFF + ph*5 + c] = d;
    }
  }
}

// ---------------------------------------------------------------------------
// K1: 24-channel conv bank via v_wmma_f32_16x16x32_bf16
//     LDS tile filled with GLOBAL_LOAD_ASYNC_TO_LDS_B32 (ASYNCcnt path).
//     block: 256 threads (8 waves); tile 64x8; wave w -> row y0+w,
//     4 groups of 16 same-phase pixels (stride-2 in x).
// ---------------------------------------------------------------------------
__constant__ int OUT0C[16] = {7,8,9,10,12,13,14,15,43,19,20,21,22,-1,41,42};
__constant__ int OUT1C[16] = {44,45,46,-2,-3,-4,-5,-6,-9,-9,-9,-9,-9,-9,-9,-9};

__global__ __launch_bounds__(256) void conv_wmma(const float* __restrict__ in,
                                                 float* __restrict__ out,
                                                 float* __restrict__ ws) {
  __shared__ float S[12*68];
  const int bz  = blockIdx.z;
  const int tx0 = blockIdx.x * 64;
  const int y0  = blockIdx.y * 8;
  const int tid = threadIdx.x;
  const float* img = in + (size_t)bz * NPIX;

  // async DMA of the (reflect-padded) bayer tile into LDS: per-lane GVS mode,
  // saddr = image base, vaddr = byte offset, vdst = LDS byte address.
  for (int i = tid; i < 12*68; i += 256) {
    int r = i / 68, c = i % 68;
    unsigned voff = ((unsigned)refl(y0 + r - 2, HW) * HW +
                     (unsigned)refl(tx0 + c - 2, HW)) * 4u;
    unsigned ldsa = (unsigned)(uintptr_t)(&S[i]);
    asm volatile("global_load_async_to_lds_b32 %0, %1, %2"
                 :: "v"(ldsa), "v"(voff), "s"(img) : "memory");
  }
  asm volatile("s_wait_asynccnt 0x0" ::: "memory");
  __syncthreads();

  const int w    = tid >> 5;
  const int lane = tid & 31;
  const int y    = y0 + w;
  const int m    = lane & 15;
  const int hi   = lane >> 4;
  const int n    = lane & 15;
  const int bm   = hi * 8;

  // preload A matrices (bank0 + bank1 for the two column phases of this row)
  v16bf a0, a1e, a1o;
  const int phE = (y & 1) * 2 + 0;
  const int phO = (y & 1) * 2 + 1;
#pragma unroll
  for (int v = 0; v < 8; ++v)
#pragma unroll
    for (int h = 0; h < 2; ++h) {
      int k = ((v >> 2) * 16) + 2*(v & 3) + h + hi*8;   // A 16x32 bf16 layout
      a0 [2*v+h] = (__bf16)ws[WB0_OFF + m*32 + k];
      a1e[2*v+h] = (__bf16)ws[WB1_OFF + phE*512 + m*32 + k];
      a1o[2*v+h] = (__bf16)ws[WB1_OFF + phO*512 + m*32 + k];
    }

  // hoist the per-lane tap -> LDS offset map (independent of pixel group)
  int  soff[16];
  bool svld[16];
#pragma unroll
  for (int v = 0; v < 8; ++v)
#pragma unroll
    for (int h = 0; h < 2; ++h) {
      int k = ((v >> 2) * 16) + 2*(v & 3) + h + hi*8;   // B 32x16 bf16 layout
      svld[2*v+h] = (k < 25);
      int dy = (k < 25) ? (k/5 - 2) : 0;
      int dx = (k < 25) ? (k%5 - 2) : 0;
      soff[2*v+h] = (w + 2 + dy)*68 + (2*n + dx + 2);
    }

#pragma unroll
  for (int g = 0; g < 4; ++g) {
    const int cb = (g >> 1) * 32 + (g & 1);
    v16bf bmat;
#pragma unroll
    for (int j = 0; j < 16; ++j) {
      float val = svld[j] ? S[soff[j] + cb] : 0.f;
      bmat[j] = (__bf16)val;
    }

    v8f c0 = {};
    v8f c1 = {};
    c0 = __builtin_amdgcn_wmma_f32_16x16x32_bf16(false, a0, false, bmat,
                                                 (short)0, c0, false, false);
    v16bf a1 = (g & 1) ? a1o : a1e;
    c1 = __builtin_amdgcn_wmma_f32_16x16x32_bf16(false, a1, false, bmat,
                                                 (short)0, c1, false, false);

    const int x = tx0 + cb + 2*n;
    const size_t pix = (size_t)y * HW + x;
#pragma unroll
    for (int i = 0; i < 8; ++i) {
      int ch = OUT0C[bm + i];
      if (ch >= 0)
        out[(((size_t)bz*52 + ch) * HW) * HW + pix] = c0[i];
      else  // mhcf -> workspace field 0
        ws[FLD_OFF + ((size_t)(0*NBAT + bz)) * NPIX + pix] = c0[i];
      int c2 = OUT1C[bm + i];
      if (c2 >= 0)
        out[(((size_t)bz*52 + c2) * HW) * HW + pix] = c1[i];
      else if (c2 >= -6) {
        int f = -c2 - 1;      // num fields 1..5
        ws[FLD_OFF + ((size_t)(f*NBAT + bz)) * NPIX + pix] = c1[i];
      }
    }
  }
}

// ---------------------------------------------------------------------------
// K2: fused elementwise / box-filter / morphology pass (remaining 34 channels)
//     reads K1 conv fields back (192MB L2 keeps them resident).
// ---------------------------------------------------------------------------
__constant__ int FCHC[7] = {12,13,21,22,43,41,42};  // gx,gy,hgh,hgv,chk,stx,sty
#define FGX 0
#define FGY 1
#define FHGH 2
#define FHGV 3
#define FCHK 4
#define FSTX 5
#define FSTY 6

__global__ __launch_bounds__(256) void fuse_feats(const float* __restrict__ in,
                                                  float* __restrict__ out,
                                                  const float* __restrict__ ws) {
  __shared__ float sB[20*20];
  __shared__ float sF[7][18*18];
  const int bz = blockIdx.z;
  const int x0 = blockIdx.x * 16, y0 = blockIdx.y * 16;
  const int tid = threadIdx.y * 16 + threadIdx.x;
  const float* img = in + (size_t)bz * NPIX;

  for (int i = tid; i < 400; i += 256) {
    int r = i / 20, c = i % 20;
    sB[i] = img[(size_t)refl(y0-2+r, HW) * HW + refl(x0-2+c, HW)];
  }
  for (int i = tid; i < 7*324; i += 256) {
    int f = i / 324, j = i % 324, r = j / 18, c = j % 18;
    const float* chp = out + (((size_t)bz*52 + FCHC[f]) * HW) * HW;
    sF[f][j] = chp[(size_t)refl(y0-1+r, HW) * HW + refl(x0-1+c, HW)];
  }
  __syncthreads();

  const int lx = threadIdx.x, ly = threadIdx.y;
  const int x = x0 + lx, y = y0 + ly;
  const size_t pix = (size_t)y * HW + x;
#define BV(dy,dx) sB[(ly+2+(dy))*20 + (lx+2+(dx))]
#define FV(f,dy,dx) sF[f][(ly+1+(dy))*18 + (lx+1+(dx))]
  auto W = [&](int ch, float v) {
    out[(((size_t)bz*52 + ch) * HW) * HW + pix] = v;
  };

  const int ry = y & 1, rx = x & 1;
  const float gbm = (ry==0 && rx==0) ? 1.f : 0.f;
  const float bmk = (ry==0 && rx==1) ? 1.f : 0.f;
  const float rmk = (ry==1 && rx==0) ? 1.f : 0.f;
  const float grm = (ry==1 && rx==1) ? 1.f : 0.f;
  const float gmk = gbm + grm;
  W(0, rmk); W(1, gmk); W(2, bmk); W(3, grm); W(4, gbm);
  W(5, (float)ry); W(6, (float)rx);

  const float bc = BV(0,0);

  // 3x3 morphological gradient + all 3x3 box sums in one neighborhood sweep
  float mx = -1e30f, mn = 1e30f;
  float jxx = 0.f, jyy = 0.f, jxy = 0.f;
  float se_chk = 0.f, se_str = 0.f, sb = 0.f, sb2 = 0.f, reh = 0.f, rev = 0.f;
#pragma unroll
  for (int dy = -1; dy <= 1; ++dy)
#pragma unroll
    for (int dx = -1; dx <= 1; ++dx) {
      float bv = BV(dy,dx);
      mx = fmaxf(mx, bv); mn = fminf(mn, bv);
      sb += bv; sb2 += bv*bv;
      float gxv = FV(FGX,dy,dx), gyv = FV(FGY,dy,dx);
      jxx += gxv*gxv; jyy += gyv*gyv; jxy += gxv*gyv;
      float ck = FV(FCHK,dy,dx); se_chk += ck*ck;
      float sx = FV(FSTX,dy,dx), sy = FV(FSTY,dy,dx); se_str += sx*sx + sy*sy;
      float rh = bv - FV(FHGH,dy,dx); reh += rh*rh;
      float rv = bv - FV(FHGV,dy,dx); rev += rv*rv;
    }
  W(11, mx - mn);

  const float gx = FV(FGX,0,0), gy = FV(FGY,0,0);
  W(16, sqrtf(gx*gx + gy*gy + EPSF));

  const float inv9 = 1.f/9.f;
  jxx *= inv9; jyy *= inv9; jxy *= inv9;
  const float tr = jxx + jyy;
  const float lam = sqrtf((jxx-jyy)*(jxx-jyy) + 4.f*jxy*jxy + EPSF);
  W(17, lam / (tr + EPSF));
  W(18, (jxx - jyy) / (tr + EPSF));

  const float hgh = FV(FHGH,0,0), hgv = FV(FHGV,0,0);
  W(23, fabsf(hgh - hgv));
  W(24, bc - hgh);
  W(25, bc - hgv);
  W(26, reh * inv9);
  W(27, rev * inv9);
  W(28, fabsf(gx) - fabsf(gy));

  const float hxx = out[(((size_t)bz*52 + 8) * HW) * HW + pix];
  const float hyy = out[(((size_t)bz*52 + 9) * HW) * HW + pix];
  W(29, fabsf(hxx) - fabsf(hyy));

  // 1x5 / 5x1 line variances
  float mh = 0.f, qh = 0.f, mv = 0.f, qv = 0.f;
#pragma unroll
  for (int d = -2; d <= 2; ++d) {
    float vh = BV(0,d); mh += vh; qh += vh*vh;
    float vv = BV(d,0); mv += vv; qv += vv*vv;
  }
  mh *= 0.2f; qh *= 0.2f; mv *= 0.2f; qv *= 0.2f;
  const float lvh = fmaxf(qh - mh*mh, 0.f);
  const float lvv = fmaxf(qv - mv*mv, 0.f);
  W(30, lvh); W(31, lvv);
  const float lvd = lvh - lvv;
  W(32, lvd);
  W(33, fabsf(lvd) / (lvh + lvv + EPSF));

  // normalized sparse fills (numerators from WMMA pass, parity denominators)
  const int ph = ry*2 + rx;
  float fills[5];
#pragma unroll
  for (int c = 0; c < 5; ++c) {
    float num = ws[FLD_OFF + ((size_t)((c+1)*NBAT + bz)) * NPIX + pix];
    float den = fmaxf(ws[DEN_OFF + ph*5 + c], EPSF);
    fills[c] = num / den;
  }
  W(34, fills[0] - fills[1]);   // rg
  W(35, fills[2] - fills[1]);   // bg
  W(36, fills[3] - fills[4]);   // gpd

  const float mhcf = ws[FLD_OFF + ((size_t)(0*NBAT + bz)) * NPIX + pix];
  const float mhc  = (gmk > 0.5f) ? bc : mhcf;
  W(37, mhc);
  W(38, mhc - 0.5f*(hgh + hgv));
  W(39, (bc - mhc) * rmk);
  W(40, (bc - mhc) * bmk);

  W(47, se_chk * inv9);
  W(48, se_str * inv9);
  const float lmean = sb * inv9;
  W(49, lmean);
  W(50, fmaxf(sb2 * inv9 - lmean*lmean, 0.f));
  W(51, tr);                    // gen == avg3(gx^2+gy^2) == jxx+jyy
#undef BV
#undef FV
}

// ---------------------------------------------------------------------------
extern "C" void kernel_launch(void* const* d_in, const int* in_sizes, int n_in,
                              void* d_out, int out_size, void* d_ws, size_t ws_size,
                              hipStream_t stream) {
  (void)in_sizes; (void)n_in; (void)out_size; (void)ws_size;
  const float* bayer = (const float*)d_in[0];
  float* out = (float*)d_out;
  float* ws  = (float*)d_ws;

  bf_init<<<1, 1, 0, stream>>>(ws);
  conv_wmma<<<dim3(HW/64, HW/8, NBAT), 256, 0, stream>>>(bayer, out, ws);
  fuse_feats<<<dim3(HW/16, HW/16, NBAT), dim3(16,16), 0, stream>>>(bayer, out, ws);
}
